// APPNP_77575699301005
// MI455X (gfx1250) — compile-verified, matrix operand order checked by reading, not run on hardware
//
#include <hip/hip_runtime.h>
#include <hip/hip_bf16.h>

typedef __attribute__((ext_vector_type(16))) _Float16 v16h;
typedef __attribute__((ext_vector_type(8)))  _Float16 v8h;
typedef __attribute__((ext_vector_type(8)))  float    v8f;
typedef __attribute__((ext_vector_type(4)))  float    v4f;

#define APPNP_ALPHA 0.1f
#define APPNP_K     10
#define NCLASS      40

// ---------------------------------------------------------------------------
// Unconditional vectorized 8-element loaders (16B global_load_b128 traffic).
// ---------------------------------------------------------------------------
__device__ __forceinline__ void load8(const float* __restrict__ p, v16h& f, int o) {
  const v4f a = *(const v4f*)p;
  const v4f b = *(const v4f*)(p + 4);
#pragma unroll
  for (int i = 0; i < 4; ++i) {
    f[o + i]     = (_Float16)a[i];
    f[o + 4 + i] = (_Float16)b[i];
  }
}

__device__ __forceinline__ void load8(const _Float16* __restrict__ p, v16h& f, int o) {
  const v8h a = *(const v8h*)p;
#pragma unroll
  for (int i = 0; i < 8; ++i) f[o + i] = a[i];
}

// A-operand layout (16-bit A 16x32, ISA 7.12.2): lane = M; lane-half selects
// the K sub-block: elems 0..7 -> k0+half*8+e, elems 8..15 -> k0+16+half*8+e.
// B mirrors A with lane = N column.
template <typename T>
__device__ __forceinline__ v16h load_frag_full(const T* __restrict__ base,
                                               int k0, int half) {
  v16h f;
  load8(base + k0 + half * 8, f, 0);
  load8(base + k0 + 16 + half * 8, f, 8);
  return f;
}

// A remainder chunk: clamp address (always-loadable), select value (v_cndmask).
template <typename T>
__device__ __forceinline__ v16h load_frag_rem(const T* __restrict__ base,
                                              int k0, int half, int K) {
  v16h f;
#pragma unroll
  for (int e = 0; e < 8; ++e) {
    const int kLo = k0 + half * 8 + e;
    const int kHi = k0 + 16 + half * 8 + e;
    const _Float16 vLo = (_Float16)base[kLo < K ? kLo : K - 1];
    const _Float16 vHi = (_Float16)base[kHi < K ? kHi : K - 1];
    f[e]     = (kLo < K) ? vLo : (_Float16)0.0f;
    f[e + 8] = (kHi < K) ? vHi : (_Float16)0.0f;
  }
  return f;
}

// ---------------------------------------------------------------------------
// One-time weight conversion f32 -> f16, rows zero-padded to ldw (mult. of 32)
// so every B fragment load is a full, 16B-aligned v8h load with no remainder.
// ---------------------------------------------------------------------------
__global__ void cvt_weights(const float* __restrict__ src, _Float16* __restrict__ dst,
                            int K, int ldw, int total) {
  const int i = blockIdx.x * blockDim.x + threadIdx.x;
  if (i >= total) return;
  const int n = i / ldw;
  const int k = i - n * ldw;
  dst[i] = (k < K) ? (_Float16)src[(size_t)n * K + k] : (_Float16)0.0f;
}

// ---------------------------------------------------------------------------
// GEMM: out[M x nOut] = relu(A[M x K] * W[nOut x K]^T + bias), f32 accumulate.
// One wave computes a 16 x (COLS*16) output strip (A fragment reused COLS x).
// Tile guard is wave-uniform so EXEC is all-ones at every WMMA.
// ---------------------------------------------------------------------------
template <typename InT, typename OutT, int COLS>
__global__ void __launch_bounds__(256)
gemm_bias_relu_wmma(const InT* __restrict__ A, int lda,
                    const _Float16* __restrict__ W16, int ldw,
                    const float* __restrict__ bias,
                    OutT* __restrict__ out, int ldo,
                    int M, int K, int nOut, int colGroups) {
  const int wave = (blockIdx.x * blockDim.x + threadIdx.x) >> 5;
  const int nTilesM = M >> 4;
  if (wave >= nTilesM * colGroups) return;   // uniform per wave

  const int mTile = wave / colGroups;
  const int cg    = wave % colGroups;
  const int lane  = threadIdx.x & 31;
  const int half  = lane >> 4;
  const int l16   = lane & 15;

  const InT* aRow = A + (size_t)(mTile * 16 + l16) * lda;
  const int nBase = cg * (COLS * 16);

  // Clamp out-of-range B columns to a valid row: their accumulator columns
  // receive garbage but are never stored (D[:,n] depends only on B[:,n]).
  const _Float16* wRows[COLS];
#pragma unroll
  for (int c = 0; c < COLS; ++c) {
    int n = nBase + c * 16 + l16;
    n = (n < nOut) ? n : (nOut - 1);
    wRows[c] = W16 + (size_t)n * ldw;
  }

  v8f acc[COLS];
#pragma unroll
  for (int c = 0; c < COLS; ++c)
#pragma unroll
    for (int i = 0; i < 8; ++i) acc[c][i] = 0.0f;

  const int kFull = K >> 5;
  for (int kc = 0; kc < kFull; ++kc) {
    const int k0 = kc << 5;
    if (k0 + 64 < K) __builtin_prefetch(aRow + k0 + 64, 0, 0);  // global_prefetch_b8
    const v16h a = load_frag_full(aRow, k0, half);
#pragma unroll
    for (int c = 0; c < COLS; ++c) {
      const v16h b = load_frag_full(wRows[c], k0, half);
      // 8 args: (neg_a, A, neg_b, B, c_mod, C, reuse_a, reuse_b)
      acc[c] = __builtin_amdgcn_wmma_f32_16x16x32_f16(
          false, a, false, b, (short)0, acc[c], false, false);
    }
  }
  if (K & 31) {
    const int k0 = kFull << 5;
    // A: clamped+masked loads; B: plain full load (padding region is zeros,
    // so zero * A-garbage contributes nothing to the accumulator).
    const v16h a = load_frag_rem(aRow, k0, half, K);
#pragma unroll
    for (int c = 0; c < COLS; ++c) {
      const v16h b = load_frag_full(wRows[c], k0, half);
      acc[c] = __builtin_amdgcn_wmma_f32_16x16x32_f16(
          false, a, false, b, (short)0, acc[c], false, false);
    }
  }

  // C/D layout: VGPR i -> M = i + 8*half, N = lane&15
#pragma unroll
  for (int c = 0; c < COLS; ++c) {
    const int n = nBase + c * 16 + l16;
    if (n < nOut) {
      const float bv = bias[n];
#pragma unroll
      for (int i = 0; i < 8; ++i) {
        const int m = mTile * 16 + half * 8 + i;
        float v = acc[c][i] + bv;
        v = v > 0.0f ? v : 0.0f;
        out[(size_t)m * ldo + n] = (OutT)v;
      }
    }
  }
}

// ---------------------------------------------------------------------------
// Degree normalization: deg = 1 (self loop) + in-degree; dinv = rsqrt(deg)
// ---------------------------------------------------------------------------
__global__ void deg_init(int* __restrict__ deg, int n) {
  const int i = blockIdx.x * blockDim.x + threadIdx.x;
  if (i < n) deg[i] = 1;
}

__global__ void deg_count(const int* __restrict__ dst, int* __restrict__ deg, int e) {
  const int i = blockIdx.x * blockDim.x + threadIdx.x;
  if (i < e) atomicAdd(&deg[dst[i]], 1);
}

__global__ void deg_rsqrt(const int* __restrict__ deg, float* __restrict__ dinv, int n) {
  const int i = blockIdx.x * blockDim.x + threadIdx.x;
  if (i < n) dinv[i] = rsqrtf((float)deg[i]);
}

// ---------------------------------------------------------------------------
// APPNP iteration, split into:
//   init:  hnext = alpha*h0 + (1-alpha)*dinv[i]^2*hcur   (self-loop + teleport)
//   edges: hnext[dst] += (1-alpha)*dinv[src]*dinv[dst]*hcur[src]  (real edges)
// ---------------------------------------------------------------------------
__global__ void prop_init(const float* __restrict__ h0, const float* __restrict__ hcur,
                          const float* __restrict__ dinv, float* __restrict__ hnext,
                          int total) {
  const int idx = blockIdx.x * blockDim.x + threadIdx.x;
  if (idx >= total) return;
  const int node = idx / NCLASS;
  const float d = dinv[node];
  hnext[idx] = APPNP_ALPHA * h0[idx] + (1.0f - APPNP_ALPHA) * d * d * hcur[idx];
}

__global__ void prop_edges(const int* __restrict__ src, const int* __restrict__ dst,
                           const float* __restrict__ dinv,
                           const float* __restrict__ hcur, float* __restrict__ hnext,
                           int E) {
  const int t = blockIdx.x * blockDim.x + threadIdx.x;
  const int e = t >> 3;                 // 8 threads per edge
  if (e >= E) return;
  const int c0 = (t & 7) * 5;           // 5 contiguous channels each (8*5 = 40)
  const int s = src[e];
  const int d = dst[e];
  const float w = (1.0f - APPNP_ALPHA) * dinv[s] * dinv[d];
  const float* __restrict__ hs = hcur + (size_t)s * NCLASS + c0;
  float* __restrict__ hd = hnext + (size_t)d * NCLASS + c0;
#pragma unroll
  for (int c = 0; c < 5; ++c) atomicAdd(hd + c, w * hs[c]);
}

// ---------------------------------------------------------------------------
// Host launcher
// ---------------------------------------------------------------------------
extern "C" void kernel_launch(void* const* d_in, const int* in_sizes, int n_in,
                              void* d_out, int out_size, void* d_ws, size_t ws_size,
                              hipStream_t stream) {
  const float* x  = (const float*)d_in[0];
  const int*   ei = (const int*)d_in[1];
  const float* W0 = (const float*)d_in[2];
  const float* b0 = (const float*)d_in[3];
  const float* W1 = (const float*)d_in[4];
  const float* b1 = (const float*)d_in[5];
  const float* W2 = (const float*)d_in[6];
  const float* b2 = (const float*)d_in[7];

  const int NFEAT = 500, H0 = 256, H1 = 128;
  const int N = in_sizes[0] / NFEAT;   // 100000 (divisible by 16)
  const int E = in_sizes[1] / 2;       // 1600000
  const int* src = ei;
  const int* dst = ei + E;

  const int ldw0 = (NFEAT + 31) & ~31;  // 512
  const int ldw1 = H0;                  // 256 (already mult of 32)
  const int ldw2 = H1;                  // 128

  // Workspace carve-out (~126 MB), 256B aligned slices.
  char* ws = (char*)d_ws;
  size_t off = 0;
  auto carve = [&](size_t bytes) -> void* {
    void* p = ws + off;
    off = (off + bytes + 255) & ~(size_t)255;
    return p;
  };
  _Float16* h1  = (_Float16*)carve((size_t)N * H0 * sizeof(_Float16));
  _Float16* h2  = (_Float16*)carve((size_t)N * H1 * sizeof(_Float16));
  float*    h0  = (float*)carve((size_t)N * NCLASS * sizeof(float));
  float*    hA  = (float*)carve((size_t)N * NCLASS * sizeof(float));
  float*    hB  = (float*)carve((size_t)N * NCLASS * sizeof(float));
  int*      deg = (int*)carve((size_t)N * sizeof(int));
  float*    dnv = (float*)carve((size_t)N * sizeof(float));
  _Float16* W0h = (_Float16*)carve((size_t)H0 * ldw0 * sizeof(_Float16));
  _Float16* W1h = (_Float16*)carve((size_t)H1 * ldw1 * sizeof(_Float16));
  _Float16* W2h = (_Float16*)carve((size_t)NCLASS * ldw2 * sizeof(_Float16));

  // --- One-time weight conversion (tiny) ---
  {
    const int t0 = H0 * ldw0, t1 = H1 * ldw1, t2 = NCLASS * ldw2;
    cvt_weights<<<(t0 + 255) / 256, 256, 0, stream>>>(W0, W0h, NFEAT, ldw0, t0);
    cvt_weights<<<(t1 + 255) / 256, 256, 0, stream>>>(W1, W1h, H0, ldw1, t1);
    cvt_weights<<<(t2 + 255) / 256, 256, 0, stream>>>(W2, W2h, H1, ldw2, t2);
  }

  const int mTiles = N / 16;

  // --- MLP encoder (WMMA f16, f32 accumulate, fused bias+ReLU) ---
  {
    const int cg = H0 / 64;  // 4
    const int tiles = mTiles * cg;
    gemm_bias_relu_wmma<float, _Float16, 4>
        <<<(tiles + 7) / 8, 256, 0, stream>>>(x, NFEAT, W0h, ldw0, b0, h1, H0,
                                              N, NFEAT, H0, cg);
  }
  {
    const int cg = H1 / 64;  // 2
    const int tiles = mTiles * cg;
    gemm_bias_relu_wmma<_Float16, _Float16, 4>
        <<<(tiles + 7) / 8, 256, 0, stream>>>(h1, H0, W1h, ldw1, b1, h2, H1,
                                              N, H0, H1, cg);
  }
  {
    const int cg = 1;        // covers n 0..63, guards trim to 40
    const int tiles = mTiles * cg;
    gemm_bias_relu_wmma<_Float16, float, 4>
        <<<(tiles + 7) / 8, 256, 0, stream>>>(h2, H1, W2h, ldw2, b2, h0, NCLASS,
                                              N, H1, NCLASS, cg);
  }

  // --- GCN degree normalization ---
  deg_init <<<(N + 255) / 256, 256, 0, stream>>>(deg, N);
  deg_count<<<(E + 255) / 256, 256, 0, stream>>>(dst, deg, E);
  deg_rsqrt<<<(N + 255) / 256, 256, 0, stream>>>(deg, dnv, N);

  // --- APPNP propagation (ping-pong, last iteration writes d_out) ---
  const int total = N * NCLASS;
  const long long edgeThreads = (long long)E * 8;
  const float* cur = h0;
  for (int it = 0; it < APPNP_K; ++it) {
    float* nxt = (it == APPNP_K - 1) ? (float*)d_out : ((it & 1) ? hB : hA);
    prop_init <<<(total + 255) / 256, 256, 0, stream>>>(h0, cur, dnv, nxt, total);
    prop_edges<<<(int)((edgeThreads + 255) / 256), 256, 0, stream>>>(src, dst, dnv,
                                                                     cur, nxt, E);
    cur = nxt;
  }
}
